// VQVAELayer_41240275976611
// MI455X (gfx1250) — matrix operand order repaired and checked by reading, not compiled, not run on hardware
//
#include <hip/hip_runtime.h>
#include <hip/hip_bf16.h>
#include <stdint.h>

// Problem sizes (fixed by the reference)
#define N_ROWS 32768   // 32*32*32
#define DIM    256
#define K_CODE 8192
#define KT_N   (K_CODE / 16)    // 512 k-tiles of 16 codes
#define NSTAGE (KT_N / 2)       // 256 stages of 32 codes (2 k-tiles)

typedef __attribute__((ext_vector_type(16))) __bf16 bf16x16;
typedef __attribute__((ext_vector_type(8)))  float  f32x8;
typedef __attribute__((ext_vector_type(4)))  int    v4i;

union AFrag { bf16x16 v; unsigned short u[16]; };
union BFrag { bf16x16 v; float4 f[2]; };

#define AS1 __attribute__((address_space(1)))
#define AS3 __attribute__((address_space(3)))

// CDNA5 async global->LDS copies (ASYNCcnt-tracked), if the toolchain has them
#if defined(__has_builtin)
#if __has_builtin(__builtin_amdgcn_global_load_async_to_lds_b128)
#define HAVE_ASYNC_LDS 1
#endif
#endif
#ifndef HAVE_ASYNC_LDS
#define HAVE_ASYNC_LDS 0
#endif

__device__ __forceinline__ void wait_asynccnt0() {
#if defined(__has_builtin)
#if __has_builtin(__builtin_amdgcn_s_wait_asynccnt)
  __builtin_amdgcn_s_wait_asynccnt(0);
  return;
#else
  asm volatile("s_wait_asynccnt 0x0" ::: "memory");
  return;
#endif
#else
  asm volatile("s_wait_asynccnt 0x0" ::: "memory");
#endif
}

__device__ __forceinline__ void async_copy_b128(const char* gsrc, char* ldst) {
  __builtin_amdgcn_global_load_async_to_lds_b128(
      (AS1 v4i*)gsrc, (AS3 v4i*)ldst, 0, 0);
}

// round-to-nearest-even f32 -> bf16 bits (as f32 bit pattern)
__device__ __forceinline__ unsigned bf16_rne_bits32(float f) {
  unsigned u = __float_as_uint(f);
  u += 0x7FFFu + ((u >> 16) & 1u);
  return u & 0xFFFF0000u;
}

// ---------------------------------------------------------------------------
// bias[k] = -0.5 * sum_d w[d][k]^2
__global__ __launch_bounds__(256) void prep_bias_kernel(const float* __restrict__ w,
                                                        float* __restrict__ bias) {
  int k = blockIdx.x * 256 + threadIdx.x;
  float s = 0.f;
  for (int d = 0; d < DIM; ++d) {
    float v = w[(size_t)d * K_CODE + k];
    s += v * v;
  }
  bias[k] = -0.5f * s;
}

// wT[k][d] = w[d][k]
__global__ __launch_bounds__(256) void prep_wT_kernel(const float* __restrict__ w,
                                                      float* __restrict__ wT) {
  size_t t = (size_t)blockIdx.x * 256 + threadIdx.x;   // over DIM*K_CODE
  int d = (int)(t >> 13);
  int k = (int)(t & (K_CODE - 1));
  wT[(size_t)k * DIM + d] = w[t];
}

// Pack w into per-lane WMMA B fragments (bf16 hi and lo), LDS-friendly order.
// Fragment (kt, ds, lane): column n = lane&15, d0 = ds*32 + 16*(lane>>4),
// elements e=0..15 -> d = d0+e, code k = kt*16+n.
// float4 slot within the 8KB-per-kt tile: s = ds*64 + half*32 + lane.
__global__ __launch_bounds__(256) void prep_wpack_kernel(const float* __restrict__ w,
                                                         unsigned short* __restrict__ whi,
                                                         unsigned short* __restrict__ wlo) {
  int f = blockIdx.x * 256 + threadIdx.x;      // 131072 fragments
  int lane = f & 31;
  int ds   = (f >> 5) & 7;
  int kt   = f >> 8;
  int n  = lane & 15;
  int d0 = ds * 32 + 16 * (lane >> 4);
  int k  = kt * 16 + n;
  size_t base = ((size_t)kt * 512 + (size_t)ds * 64 + lane) * 8;  // ushort idx of e=0..7
  #pragma unroll
  for (int e = 0; e < 16; ++e) {
    float v = w[(size_t)(d0 + e) * K_CODE + k];
    unsigned hb = bf16_rne_bits32(v);
    float rem = v - __uint_as_float(hb);
    unsigned lb = bf16_rne_bits32(rem);
    size_t o = (e < 8) ? (base + e) : (base + 8 * 32 + (e - 8));
    whi[o] = (unsigned short)(hb >> 16);
    wlo[o] = (unsigned short)(lb >> 16);
  }
}

__global__ void init_acc_kernel(float* acc) { acc[0] = 0.f; }

// ---------------------------------------------------------------------------
// Core: per 16-row tile, scan all K codes with bf16x3 WMMA, track argmax of
// (x . w_k - 0.5*||w_k||^2). 4 waves / block; w tiles staged through LDS with
// async global->LDS double buffering (32KB per stage = 2 k-tiles of 16 codes).
__global__ __launch_bounds__(128) void vq_argmin_kernel(const float* __restrict__ x,
                                                        const float* __restrict__ whi_p,
                                                        const float* __restrict__ wlo_p,
                                                        const float* __restrict__ bias,
                                                        int* __restrict__ idx_out) {
#if HAVE_ASYNC_LDS
  __shared__ float4 lds4[4096];                // 2 x 32KB double buffer
#else
  __shared__ float4 lds4[2048];                // single 32KB stage buffer
#endif
  const int tid  = threadIdx.x;
  const int lane = tid & 31;
  const int wave = tid >> 5;
  const int row_base = blockIdx.x * 64 + wave * 16;

  // ---- Resident A fragments: 16 rows of x, split hi/lo bf16 ----
  // A k order used here (contiguous, same as B packing): lanes 0-15: M=lane, d = ds*32 + {0..15};
  // lanes 16-31: M=lane-16, d = ds*32 + {16..31}.
  AFrag a_hi[8], a_lo[8];
  {
    const int m = lane & 15;
    const float* xr = x + (size_t)(row_base + m) * DIM;
    const int dadd = (lane >> 4) * 16;
    #pragma unroll
    for (int ds = 0; ds < 8; ++ds) {
      const int d0 = ds * 32 + dadd;
      #pragma unroll
      for (int e = 0; e < 16; ++e) {
        int d = d0 + e;
        float v = xr[d];
        unsigned hb = bf16_rne_bits32(v);
        float rem = v - __uint_as_float(hb);
        unsigned lb = bf16_rne_bits32(rem);
        a_hi[ds].u[e] = (unsigned short)(hb >> 16);
        a_lo[ds].u[e] = (unsigned short)(lb >> 16);
      }
    }
  }

  float bestv[8];
  int   besti[8];
  #pragma unroll
  for (int j = 0; j < 8; ++j) { bestv[j] = -3.4e38f; besti[j] = 0; }

#if HAVE_ASYNC_LDS
  // Per stage: block copies 16KB of hi + 16KB of lo fragments. Per builtin
  // call: 16B per lane, lanes contiguous -> 2KB per block-wide iteration.
  const char* ghi = (const char*)whi_p;
  const char* glo = (const char*)wlo_p;
  char* lbase = (char*)lds4;
  #pragma unroll
  for (int i = 0; i < 8; ++i) {               // prefetch stage 0 into buffer 0
    async_copy_b128(ghi + i * 2048 + tid * 16,
                    lbase + i * 2048 + tid * 16);
    async_copy_b128(glo + i * 2048 + tid * 16,
                    lbase + 16384 + i * 2048 + tid * 16);
  }
#else
  const float4* whi4 = (const float4*)whi_p;
  const float4* wlo4 = (const float4*)wlo_p;
  float4 r[16];
  #pragma unroll
  for (int j = 0; j < 8; ++j) {
    r[j]     = whi4[(size_t)tid * 8 + j];
    r[8 + j] = wlo4[(size_t)tid * 8 + j];
  }
#endif

  int buf = 0;
  for (int s = 0; s < NSTAGE; ++s) {
#if HAVE_ASYNC_LDS
    wait_asynccnt0();                          // this wave's copies landed
    __syncthreads();                           // everyone's copies visible;
                                               // prev buffer fully consumed
    if (s + 1 < NSTAGE) {                      // uniform branch
      const int nb = buf ^ 1;
      #pragma unroll
      for (int i = 0; i < 8; ++i) {
        async_copy_b128(ghi + (size_t)(s + 1) * 32768 + i * 2048 + tid * 16,
                        lbase + nb * 32768 + i * 2048 + tid * 16);
        async_copy_b128(glo + (size_t)(s + 1) * 32768 + i * 2048 + tid * 16,
                        lbase + nb * 32768 + 16384 + i * 2048 + tid * 16);
      }
    }
    const int base = buf * 2048;               // float4 index of current buffer
#else
    __syncthreads();
    #pragma unroll
    for (int j = 0; j < 8; ++j) {
      lds4[tid * 8 + j]        = r[j];
      lds4[1024 + tid * 8 + j] = r[8 + j];
    }
    __syncthreads();
    if (s + 1 < NSTAGE) {
      #pragma unroll
      for (int j = 0; j < 8; ++j) {
        r[j]     = whi4[(size_t)(s + 1) * 1024 + tid * 8 + j];
        r[8 + j] = wlo4[(size_t)(s + 1) * 1024 + tid * 8 + j];
      }
    }
    const int base = 0;
#endif

    #pragma unroll
    for (int t = 0; t < 2; ++t) {              // two 16-code k-tiles per stage
      const int kt = s * 2 + t;
      const float bias_k = bias[kt * 16 + (lane & 15)];
      f32x8 c0 = {0.f, 0.f, 0.f, 0.f, 0.f, 0.f, 0.f, 0.f};  // hi*hi chain
      f32x8 c1 = {0.f, 0.f, 0.f, 0.f, 0.f, 0.f, 0.f, 0.f};  // cross-term chain
      #pragma unroll
      for (int ds = 0; ds < 8; ++ds) {
        BFrag bh, bl;
        bh.f[0] = lds4[base + t * 512 + ds * 64 + lane];
        bh.f[1] = lds4[base + t * 512 + ds * 64 + 32 + lane];
        bl.f[0] = lds4[base + 1024 + t * 512 + ds * 64 + lane];
        bl.f[1] = lds4[base + 1024 + t * 512 + ds * 64 + 32 + lane];
        c0 = __builtin_amdgcn_wmma_f32_16x16x32_bf16(false, a_hi[ds].v, false, bh.v,
                                                     (short)0, c0, false, false);
        c1 = __builtin_amdgcn_wmma_f32_16x16x32_bf16(false, a_lo[ds].v, false, bh.v,
                                                     (short)0, c1, false, false);
        c1 = __builtin_amdgcn_wmma_f32_16x16x32_bf16(false, a_hi[ds].v, false, bl.v,
                                                     (short)0, c1, false, false);
      }
      const int k = kt * 16 + (lane & 15);
      #pragma unroll
      for (int j = 0; j < 8; ++j) {
        float sc = c0[j] + c1[j] + bias_k;     // strict '>' keeps lowest k on ties
        if (sc > bestv[j]) { bestv[j] = sc; besti[j] = k; }
      }
    }
    buf ^= 1;
  }

  // Cross-lane argmax within each 16-lane group (rows 0-7 in lanes 0-15,
  // rows 8-15 in lanes 16-31); tie-break toward the smaller index.
  #pragma unroll
  for (int j = 0; j < 8; ++j) {
    #pragma unroll
    for (int off = 1; off < 16; off <<= 1) {
      float ov = __shfl_xor(bestv[j], off, 32);
      int   oi = __shfl_xor(besti[j], off, 32);
      if (ov > bestv[j] || (ov == bestv[j] && oi < besti[j])) {
        bestv[j] = ov; besti[j] = oi;
      }
    }
  }
  if ((lane & 15) == 0) {
    int rb = row_base + (lane >> 4) * 8;
    #pragma unroll
    for (int j = 0; j < 8; ++j) idx_out[rb + j] = besti[j];
  }
}

// ---------------------------------------------------------------------------
// Gather winning codebook rows (contiguous via wT), write out, reduce loss.
__global__ __launch_bounds__(256) void gather_loss_kernel(const float* __restrict__ x,
                                                          const float* __restrict__ wT,
                                                          const int* __restrict__ idx,
                                                          float* __restrict__ out,
                                                          float* __restrict__ acc) {
  __shared__ float red[256];
  int n = blockIdx.x;
  int d = threadIdx.x;
  int k = idx[n];
  float q  = wT[(size_t)k * DIM + d];
  float xv = x[(size_t)n * DIM + d];
  out[(size_t)n * DIM + d] = q;
  float diff = q - xv;
  red[d] = diff * diff;
  __syncthreads();
  #pragma unroll
  for (int s = 128; s > 0; s >>= 1) {
    if (d < s) red[d] += red[d + s];
    __syncthreads();
  }
  if (d == 0) atomicAdd(acc, red[0]);
}

__global__ void finalize_kernel(const float* acc, float* out_loss) {
  out_loss[0] = 2.0f * acc[0] / (float)((size_t)N_ROWS * DIM);  // e_loss + q_loss
}

// ---------------------------------------------------------------------------
extern "C" void kernel_launch(void* const* d_in, const int* in_sizes, int n_in,
                              void* d_out, int out_size, void* d_ws, size_t ws_size,
                              hipStream_t stream) {
  const float* x = (const float*)d_in[0];   // [32768, 256] f32
  const float* w = (const float*)d_in[1];   // [256, 8192]  f32
  float* out = (float*)d_out;               // [32768*256] quantized, then loss scalar

  char* ws = (char*)d_ws;
  const size_t MB = 1024u * 1024u;
  unsigned short* whi = (unsigned short*)(ws);               // 4 MB packed bf16 hi
  unsigned short* wlo = (unsigned short*)(ws + 4 * MB);      // 4 MB packed bf16 lo
  float* bias = (float*)(ws + 8 * MB);                       // 32 KB
  float* wT   = (float*)(ws + 8 * MB + 32 * 1024);           // 8 MB
  int*   idx  = (int*)  (ws + 16 * MB + 32 * 1024);          // 128 KB
  float* acc  = (float*)(ws + 16 * MB + 160 * 1024);         // 4 B

  prep_bias_kernel <<<K_CODE / 256, 256, 0, stream>>>(w, bias);
  prep_wT_kernel   <<<(DIM * K_CODE) / 256, 256, 0, stream>>>(w, wT);
  prep_wpack_kernel<<<(KT_N * 8 * 32) / 256, 256, 0, stream>>>(w, whi, wlo);
  init_acc_kernel  <<<1, 1, 0, stream>>>(acc);

  vq_argmin_kernel <<<N_ROWS / 64, 128, 0, stream>>>((const float*)x,
                                                     (const float*)whi,
                                                     (const float*)wlo,
                                                     bias, idx);

  gather_loss_kernel<<<N_ROWS, 256, 0, stream>>>(x, wT, idx, out, acc);
  finalize_kernel   <<<1, 1, 0, stream>>>(acc, out + (out_size - 1));
}
